// ProposalModule_53077205844205
// MI455X (gfx1250) — compile-verified, matrix operand order checked by reading, not run on hardware
//
#include <hip/hip_runtime.h>

typedef _Float16 v16h  __attribute__((ext_vector_type(16)));
typedef float    v8f   __attribute__((ext_vector_type(8)));
typedef int      i32x4 __attribute__((ext_vector_type(4)));

#define RADIUS_F  0.3f
#define NSAMPLE   16
#define NPOINT    256
#define BN_EPS    1e-5f
#define NBATCH    32
#define NPTS      2048
#define CFEAT     256

// ---------------------------------------------------------------------------
// CDNA5 async global->LDS copy (ASYNCcnt-tracked). Builtin confirmed present;
// it takes (__global i32x4*, __local i32x4*, imm offset, imm cpol).
// ---------------------------------------------------------------------------
#if defined(__has_builtin)
#  if __has_builtin(__builtin_amdgcn_global_load_async_to_lds_b128)
#    define HAVE_ASYNC_LDS 1
#  endif
#endif

__device__ __forceinline__ void stage_w_async(const uint4* __restrict__ src,
                                              uint4* lds, int t)
{
    for (int i = t; i < 2048; i += 256) {   // 2048 * 16B = 32 KB
#ifdef HAVE_ASYNC_LDS
        __builtin_amdgcn_global_load_async_to_lds_b128(
            (__attribute__((address_space(1))) i32x4*)(const void*)(src + i),
            (__attribute__((address_space(3))) i32x4*)(void*)(lds + i),
            0, 0);
#else
        lds[i] = src[i];
#endif
    }
}

__device__ __forceinline__ void wait_async_copies()
{
#ifdef HAVE_ASYNC_LDS
#  if __has_builtin(__builtin_amdgcn_s_wait_asynccnt)
    __builtin_amdgcn_s_wait_asynccnt(0);
#  else
    asm volatile("s_wait_asynccnt 0x0" ::: "memory");
#  endif
#endif
}

// ---------------------------------------------------------------------------
// Kernel A: per-batch FPS + ball query, staged entirely in LDS.
// One block (256 threads) per batch.
// ---------------------------------------------------------------------------
__global__ __launch_bounds__(256) void fps_ball_kernel(
    const float* __restrict__ xyz,
    float* __restrict__ ws_newxyz,   // [B*NPOINT*3]
    int*   __restrict__ ws_ball,     // [B*NPOINT*NSAMPLE]
    float* __restrict__ out_newxyz,  // d_out section
    float* __restrict__ out_fps)     // d_out section (int bit-pattern)
{
    __shared__ float sx[NPTS], sy[NPTS], sz[NPTS];
    __shared__ float dists[NPTS];
    __shared__ float rval[256];
    __shared__ int   ridx[256];
    __shared__ float ncx[NPOINT], ncy[NPOINT], ncz[NPOINT];
    __shared__ int   scur;

    const int b = blockIdx.x;
    const int t = threadIdx.x;
    const float* xb = xyz + (size_t)b * NPTS * 3;

    for (int n = t; n < NPTS; n += 256) {
        sx[n] = xb[n * 3 + 0];
        sy[n] = xb[n * 3 + 1];
        sz[n] = xb[n * 3 + 2];
        dists[n] = 1e10f;
    }
    if (t == 0) scur = 0;

    for (int it = 0; it < NPOINT; ++it) {
        __syncthreads();
        const int cur = scur;
        const float lx = sx[cur], ly = sy[cur], lz = sz[cur];
        if (t == 0) {
            out_fps[b * NPOINT + it] = __int_as_float(cur);
            ncx[it] = lx; ncy[it] = ly; ncz[it] = lz;
            const int g = b * NPOINT + it;
            ws_newxyz[g * 3 + 0] = lx;  ws_newxyz[g * 3 + 1] = ly;  ws_newxyz[g * 3 + 2] = lz;
            out_newxyz[g * 3 + 0] = lx; out_newxyz[g * 3 + 1] = ly; out_newxyz[g * 3 + 2] = lz;
        }
        float bv = -1.0f; int bi = 0;
        for (int n = t; n < NPTS; n += 256) {
            const float dx = sx[n] - lx, dy = sy[n] - ly, dz = sz[n] - lz;
            const float d  = dx * dx + dy * dy + dz * dz;
            const float dm = fminf(dists[n], d);
            dists[n] = dm;
            if (dm > bv) { bv = dm; bi = n; }
        }
        rval[t] = bv; ridx[t] = bi;
        __syncthreads();
        for (int off = 128; off > 0; off >>= 1) {
            if (t < off) {
                const float v2 = rval[t + off]; const int i2 = ridx[t + off];
                if (v2 > rval[t] || (v2 == rval[t] && i2 < ridx[t])) { rval[t] = v2; ridx[t] = i2; }
            }
            __syncthreads();
        }
        if (t == 0) scur = ridx[0];
    }
    __syncthreads();

    // ball query: one thread per center; first NSAMPLE in-radius indices, pad with first hit
    {
        const float cx = ncx[t], cy = ncy[t], cz = ncz[t];
        int* ob = ws_ball + ((size_t)(b * NPOINT + t)) * NSAMPLE;
        const float r2 = RADIUS_F * RADIUS_F;
        int cnt = 0, first = 0;
        for (int n = 0; n < NPTS && cnt < NSAMPLE; ++n) {
            const float dx = sx[n] - cx, dy = sy[n] - cy, dz = sz[n] - cz;
            if (dx * dx + dy * dy + dz * dz < r2) {
                if (cnt == 0) first = n;
                ob[cnt++] = n;
            }
        }
        for (int s2 = cnt; s2 < NSAMPLE; ++s2) ob[s2] = first;
    }
}

// ---------------------------------------------------------------------------
// Kernel B: pre-swizzle the three MLP weight matrices into per-lane WMMA
// B-fragment layout (f16), K padded to a multiple of 32.
// Layout: idx(ntile,kstep,lane,v) = (((ntile*KS+kstep)*32)+lane)*8 + v
// Each entry is a u32 = {f16(k0), f16(k0+1)}.
// ---------------------------------------------------------------------------
__global__ void weight_swizzle_kernel(
    const float* __restrict__ w0, const float* __restrict__ w1, const float* __restrict__ w2,
    unsigned* __restrict__ w0s, unsigned* __restrict__ w1s, unsigned* __restrict__ w2s)
{
    const int SZ0 = 8 * 9 * 32 * 8;   // 18432
    const int SZ1 = 8 * 4 * 32 * 8;   // 8192
    int tid = blockIdx.x * 256 + threadIdx.x;

    const float* w; unsigned* dst; int KS, Cin, base;
    if (tid < SZ0)              { w = w0; dst = w0s; KS = 9; Cin = 259; base = tid; }
    else if (tid < SZ0 + SZ1)   { w = w1; dst = w1s; KS = 4; Cin = 128; base = tid - SZ0; }
    else if (tid < SZ0 + 2*SZ1) { w = w2; dst = w2s; KS = 4; Cin = 128; base = tid - SZ0 - SZ1; }
    else return;

    const int v     = base & 7;
    const int lane  = (base >> 3) & 31;
    const int idx2  = base >> 8;
    const int kstep = idx2 % KS;
    const int ntile = idx2 / KS;

    const int n2   = ntile * 16 + (lane & 15);
    const int half = lane >> 4;
    const int k0   = kstep * 32 + (v & 3) * 2 + ((v >> 2) * 16) + half * 8;

    union HU { _Float16 h; unsigned short u; };
    HU a, c;
    a.h = (k0     < Cin) ? (_Float16)w[n2 * Cin + k0]     : (_Float16)0.0f;
    c.h = (k0 + 1 < Cin) ? (_Float16)w[n2 * Cin + k0 + 1] : (_Float16)0.0f;
    dst[base] = (unsigned)a.u | ((unsigned)c.u << 16);
}

// ---------------------------------------------------------------------------
// WMMA tile GEMM: one wave computes a 16x128 output stripe (8 N-tiles).
// A fragments from LDS (two ds_load_b128 per lane); B fragments from either
// global (layer 0) or LDS (layers 1-2, async-staged) -- address-space is
// inferred from the pointer argument at each inlined call site.
// ---------------------------------------------------------------------------
__device__ __forceinline__ void gemm_tiles(
    const _Float16* __restrict__ inb, int stride, const uint4* __restrict__ wsw,
    int ksteps, int mtile, int lane, v8f acc[8])
{
    for (int i = 0; i < 8; ++i)
        for (int k = 0; k < 8; ++k) acc[i][k] = 0.0f;

    const int mrow = mtile * 16 + (lane & 15);
    const int half = lane >> 4;

    for (int ks = 0; ks < ksteps; ++ks) {
        union { v16h v; uint4 q[2]; } af;
        const uint4* pa = reinterpret_cast<const uint4*>(inb + mrow * stride + ks * 32 + half * 8);
        af.q[0] = pa[0];   // K = base+0..7   (pairs)
        af.q[1] = pa[2];   // K = base+16..23 (pairs)
        for (int nt = 0; nt < 8; ++nt) {
            union { v16h v; uint4 q[2]; } bf;
            const uint4* pb = wsw + ((size_t)((nt * ksteps + ks) * 32 + lane)) * 2;
            bf.q[0] = pb[0];
            bf.q[1] = pb[1];
            acc[nt] = __builtin_amdgcn_wmma_f32_16x16x32_f16(
                false, af.v, false, bf.v, (short)0, acc[nt], false, false);
        }
    }
}

struct MlpParams {
    const float *g0, *b0, *m0, *v0;
    const float *g1, *b1, *m1, *v1;
    const float *g2, *b2, *m2, *v2;
    const float *c1w, *c1b, *bn1g, *bn1b, *bn1m, *bn1v;
    const float *c2w, *c2b, *bn2g, *bn2b, *bn2m, *bn2v;
    const float *c3w, *c3b;
};

// ---------------------------------------------------------------------------
// Kernel C: grouped MLP (3 WMMA GEMM layers + BN/ReLU), max-pool over S=16,
// FC head. One block = 8 centers (128 rows), 8 waves; wave w owns M-tile w
// (exactly one center -> in-register max-pool). Layer-1/2 weights are
// async-staged into LDS, w1 overlapped with gather + layer 0, w2 only with BN setup.
// ---------------------------------------------------------------------------
__global__ __launch_bounds__(256) void mlp_kernel(
    const float* __restrict__ xyz, const float* __restrict__ features,
    const int* __restrict__ ws_ball, const float* __restrict__ ws_newxyz,
    const uint4* __restrict__ w0s, const uint4* __restrict__ w1s, const uint4* __restrict__ w2s,
    MlpParams P, float* __restrict__ d_out)
{
    __shared__ __align__(16) _Float16 bufA[128 * 288];
    __shared__ __align__(16) _Float16 bufB[128 * 128];
    __shared__ __align__(16) uint4    wLds[2048];        // 32 KB weight stage
    __shared__ int   pidx[128];
    __shared__ float sc[128], sh[128];
    __shared__ float feat[8 * 128];
    __shared__ float h1s[8 * 128];
    __shared__ float h2s[8 * 128];

    const int t    = threadIdx.x;
    const int lane = t & 31;
    const int wave = t >> 5;
    const int b    = blockIdx.x >> 5;
    const int cg   = blockIdx.x & 31;
    const int c0   = cg * 8;

    // kick off async stage of layer-1 weights; overlaps gather + layer-0 GEMM
    stage_w_async(w1s, wLds, t);

    // ---- stage g_xyz + padding, and layer-0 BN constants ----
    if (t < 128) {
        const int ci = t >> 4, s = t & 15;
        const int gp = b * NPOINT + c0 + ci;
        const int pi = ws_ball[(size_t)gp * NSAMPLE + s];
        pidx[t] = pi;
        const float nx = ws_newxyz[gp * 3 + 0];
        const float ny = ws_newxyz[gp * 3 + 1];
        const float nz = ws_newxyz[gp * 3 + 2];
        const float* p = xyz + ((size_t)b * NPTS + pi) * 3;
        const float inv_r = 1.0f / RADIUS_F;
        bufA[t * 288 + 0] = (_Float16)((p[0] - nx) * inv_r);
        bufA[t * 288 + 1] = (_Float16)((p[1] - ny) * inv_r);
        bufA[t * 288 + 2] = (_Float16)((p[2] - nz) * inv_r);
        for (int k = 259; k < 288; ++k) bufA[t * 288 + k] = (_Float16)0.0f;

        const float s0 = P.g0[t] * rsqrtf(P.v0[t] + BN_EPS);
        sc[t] = s0;
        sh[t] = P.b0[t] - P.m0[t] * s0;
    }
    __syncthreads();

    // ---- gather features: i indexes (channel c, row r) ----
    for (int i = t; i < CFEAT * 128; i += 256) {
        const int c = i >> 7, r = i & 127;
        bufA[r * 288 + 3 + c] = (_Float16)features[((size_t)b * CFEAT + c) * NPTS + pidx[r]];
    }
    __syncthreads();

    v8f acc[8];
    const int nbase = lane & 15;
    const int half  = lane >> 4;

    // ---- layer 0: 128x288 @ 288x128 (B from global; L2-resident) ----
    gemm_tiles(bufA, 288, w0s, 9, wave, lane, acc);
    for (int nt = 0; nt < 8; ++nt) {
        const int n2 = nt * 16 + nbase;
        const float s0 = sc[n2], h0 = sh[n2];
        for (int r = 0; r < 8; ++r) {
            const float v = fmaxf(acc[nt][r] * s0 + h0, 0.0f);
            const int m = wave * 16 + r + 8 * half;
            bufB[m * 128 + n2] = (_Float16)v;
        }
    }
    __syncthreads();
    if (t < 128) {
        const float s1 = P.g1[t] * rsqrtf(P.v1[t] + BN_EPS);
        sc[t] = s1; sh[t] = P.b1[t] - P.m1[t] * s1;
    }
    wait_async_copies();   // w1 landed in wLds (this wave's copies)
    __syncthreads();       // ...and everyone else's

    // ---- layer 1: 128x128 @ 128x128 (bufB -> bufA reused, B from LDS) ----
    gemm_tiles(bufB, 128, (const uint4*)wLds, 4, wave, lane, acc);
    for (int nt = 0; nt < 8; ++nt) {
        const int n2 = nt * 16 + nbase;
        const float s0 = sc[n2], h0 = sh[n2];
        for (int r = 0; r < 8; ++r) {
            const float v = fmaxf(acc[nt][r] * s0 + h0, 0.0f);
            const int m = wave * 16 + r + 8 * half;
            bufA[m * 128 + n2] = (_Float16)v;
        }
    }
    __syncthreads();       // all waves done reading wLds(w1) and writing bufA

    stage_w_async(w2s, wLds, t);   // overlap w2 stage with BN-const setup
    if (t < 128) {
        const float s2 = P.g2[t] * rsqrtf(P.v2[t] + BN_EPS);
        sc[t] = s2; sh[t] = P.b2[t] - P.m2[t] * s2;
    }
    wait_async_copies();
    __syncthreads();

    // ---- layer 2 + max-pool over the 16 samples of this wave's center ----
    gemm_tiles(bufA, 128, (const uint4*)wLds, 4, wave, lane, acc);
    for (int nt = 0; nt < 8; ++nt) {
        const int n2 = nt * 16 + nbase;
        const float s0 = sc[n2], h0 = sh[n2];
        float mx = -1e30f;
        for (int r = 0; r < 8; ++r)
            mx = fmaxf(mx, fmaxf(acc[nt][r] * s0 + h0, 0.0f));
        mx = fmaxf(mx, __shfl_xor(mx, 16, 32));
        if (half == 0) feat[wave * 128 + n2] = mx;
    }
    __syncthreads();

    // ---- FC1: relu(bn(feat @ c1w^T + c1b)) ----
    for (int i = t; i < 8 * 128; i += 256) {
        const int ci = i >> 7, o = i & 127;
        const float* fr = feat + ci * 128;
        const float* wr = P.c1w + o * 128;
        float a2 = 0.0f;
        for (int c = 0; c < 128; ++c) a2 += fr[c] * wr[c];
        const float s0 = P.bn1g[o] * rsqrtf(P.bn1v[o] + BN_EPS);
        h1s[ci * 128 + o] = fmaxf((a2 + P.c1b[o] - P.bn1m[o]) * s0 + P.bn1b[o], 0.0f);
    }
    __syncthreads();

    // ---- FC2 ----
    for (int i = t; i < 8 * 128; i += 256) {
        const int ci = i >> 7, o = i & 127;
        const float* fr = h1s + ci * 128;
        const float* wr = P.c2w + o * 128;
        float a2 = 0.0f;
        for (int c = 0; c < 128; ++c) a2 += fr[c] * wr[c];
        const float s0 = P.bn2g[o] * rsqrtf(P.bn2v[o] + BN_EPS);
        h2s[ci * 128 + o] = fmaxf((a2 + P.c2b[o] - P.bn2m[o]) * s0 + P.bn2b[o], 0.0f);
    }
    __syncthreads();

    // ---- head: 17 outputs, routed to d_out sections ----
    if (t < 8 * 17) {
        const int ci = t / 17, o = t % 17;
        const float* fr = h2s + ci * 128;
        const float* wr = P.c3w + o * 128;
        float a2 = 0.0f;
        for (int c = 0; c < 128; ++c) a2 += fr[c] * wr[c];
        const float net = a2 + P.c3b[o];
        const int gp = b * NPOINT + c0 + ci;
        if (o < 2)      d_out[gp * 2 + o] = net;                                             // spotness
        else if (o < 5) d_out[16384 + gp * 3 + (o - 2)] = ws_newxyz[gp * 3 + (o - 2)] + net; // center
        else            d_out[40960 + gp * 12 + (o - 5)] = net;                              // heading
    }
}

// ---------------------------------------------------------------------------
// Host launcher
// ---------------------------------------------------------------------------
extern "C" void kernel_launch(void* const* d_in, const int* in_sizes, int n_in,
                              void* d_out, int out_size, void* d_ws, size_t ws_size,
                              hipStream_t stream) {
    (void)in_sizes; (void)n_in; (void)out_size; (void)ws_size;

    const float* xyz      = (const float*)d_in[0];
    const float* features = (const float*)d_in[1];
    const float* mlp_w0   = (const float*)d_in[2];
    MlpParams P;
    P.g0 = (const float*)d_in[3];  P.b0 = (const float*)d_in[4];
    P.m0 = (const float*)d_in[5];  P.v0 = (const float*)d_in[6];
    const float* mlp_w1   = (const float*)d_in[7];
    P.g1 = (const float*)d_in[8];  P.b1 = (const float*)d_in[9];
    P.m1 = (const float*)d_in[10]; P.v1 = (const float*)d_in[11];
    const float* mlp_w2   = (const float*)d_in[12];
    P.g2 = (const float*)d_in[13]; P.b2 = (const float*)d_in[14];
    P.m2 = (const float*)d_in[15]; P.v2 = (const float*)d_in[16];
    P.c1w  = (const float*)d_in[17]; P.c1b  = (const float*)d_in[18];
    P.bn1g = (const float*)d_in[19]; P.bn1b = (const float*)d_in[20];
    P.bn1m = (const float*)d_in[21]; P.bn1v = (const float*)d_in[22];
    P.c2w  = (const float*)d_in[23]; P.c2b  = (const float*)d_in[24];
    P.bn2g = (const float*)d_in[25]; P.bn2b = (const float*)d_in[26];
    P.bn2m = (const float*)d_in[27]; P.bn2v = (const float*)d_in[28];
    P.c3w  = (const float*)d_in[29]; P.c3b  = (const float*)d_in[30];

    char* ws = (char*)d_ws;
    float*    ws_newxyz = (float*)(ws + 0);        // 24576 floats  ->  98304 B
    int*      ws_ball   = (int*)(ws + 98304);      // 131072 ints   -> 524288 B
    unsigned* w0s       = (unsigned*)(ws + 622592);// 18432 u32     ->  73728 B
    unsigned* w1s       = (unsigned*)(ws + 696320);//  8192 u32     ->  32768 B
    unsigned* w2s       = (unsigned*)(ws + 729088);//  8192 u32     ->  32768 B

    float* out        = (float*)d_out;
    float* out_newxyz = out + 139264; // after spot(16384)+center(24576)+heading(98304)
    float* out_fps    = out + 163840;

    fps_ball_kernel<<<NBATCH, 256, 0, stream>>>(xyz, ws_newxyz, ws_ball, out_newxyz, out_fps);
    weight_swizzle_kernel<<<136, 256, 0, stream>>>(mlp_w0, mlp_w1, mlp_w2, w0s, w1s, w2s);
    mlp_kernel<<<NBATCH * 32, 256, 0, stream>>>(
        xyz, features, ws_ball, ws_newxyz,
        (const uint4*)w0s, (const uint4*)w1s, (const uint4*)w2s, P, out);
}
